// Multil_head_att_en_de_76304388980802
// MI455X (gfx1250) — matrix operand order repaired and checked by reading, not compiled, not run on hardware
//
#include <hip/hip_runtime.h>
#include <hip/hip_bf16.h>

// ---------------------------------------------------------------------------
// Multi-head cross attention for MI455X (gfx1250, wave32, WMMA).
//
// Roofline: ~172 GFLOP vs ~150MB unique traffic -> compute bound -> all five
// matmuls (Q/K/V proj, QK^T, PV, out-proj) on v_wmma_f32_16x16x32_f16.
// This round: 32x64 tiles per wave (1.5 loads/WMMA instead of 2.5) and
// register double-buffering so s_wait_loadcnt no longer serializes the
// load->WMMA pipeline.
// ---------------------------------------------------------------------------

#define E_DIM  1024
#define H_NUM  16
#define HD_DIM 64
#define B_NUM  8
#define SENG   2048
#define SCHI   1024

typedef __attribute__((ext_vector_type(16))) _Float16 v16h;
typedef __attribute__((ext_vector_type(8)))  _Float16 v8h;
typedef __attribute__((ext_vector_type(4)))  _Float16 v4h;
typedef __attribute__((ext_vector_type(8)))  float    v8f;

static __device__ __forceinline__ v8f wmma_f16(v16h a, v16h b, v8f c) {
  // D = A(16x32 f16) * B(32x16 f16) + C(16x16 f32)
  return __builtin_amdgcn_wmma_f32_16x16x32_f16(false, a, false, b,
                                                (short)0, c, false, false);
}

// A-fragment (16x32, f16, row-major source, tile origin already applied).
// ISA layout: lane<16 -> M=lane, K=[0..7]+[16..23]; lane>=16 -> K=[8..15]+[24..31].
static __device__ __forceinline__ v16h ld_afrag(const _Float16* __restrict__ A,
                                                int lda, int lane) {
  const int row = lane & 15;
  const int kb  = (lane >> 4) * 8;
  const _Float16* p = A + (size_t)row * lda + kb;
  v8h lo = *(const v8h*)(p);
  v8h hi = *(const v8h*)(p + 16);
  v16h r;
#pragma unroll
  for (int i = 0; i < 8; ++i) { r[i] = lo[i]; r[i + 8] = hi[i]; }
  return r;
}

// B-fragment (32x16, f16) from a transposed "BT[N][K]" row-major source.
// ISA layout: lane<16 -> N=lane, K=[0..7]+[16..23]; lane>=16 -> K=[8..15]+[24..31].
static __device__ __forceinline__ v16h ld_bfrag(const _Float16* __restrict__ BT,
                                                int ldb, int lane) {
  const int n  = lane & 15;
  const int kb = (lane >> 4) * 8;
  const _Float16* p = BT + (size_t)n * ldb + kb;
  v8h lo = *(const v8h*)(p);
  v8h hi = *(const v8h*)(p + 16);
  v16h r;
#pragma unroll
  for (int i = 0; i < 8; ++i) { r[i] = lo[i]; r[i + 8] = hi[i]; }
  return r;
}

// ---------------------------------------------------------------------------
// Stage 1: conversions
// ---------------------------------------------------------------------------
__global__ void cvt_f32_to_f16(const float* __restrict__ in,
                               _Float16* __restrict__ out, size_t n4) {
  size_t i = ((size_t)blockIdx.x * blockDim.x + threadIdx.x);
  if (i < n4) {
    const float4 v = ((const float4*)in)[i];
    v4h p = {(_Float16)v.x, (_Float16)v.y, (_Float16)v.z, (_Float16)v.w};
    ((v4h*)out)[i] = p;
  }
}

// WT[h][c][r] = (f16) W[h][r][c]
__global__ void transpose_cvt(const float* __restrict__ W,
                              _Float16* __restrict__ WT, int R, int C) {
  int h = blockIdx.z;
  int c = blockIdx.x * 16 + threadIdx.x;
  int r = blockIdx.y * 16 + threadIdx.y;
  if (r < R && c < C)
    WT[((size_t)h * C + c) * R + r] = (_Float16)W[((size_t)h * R + r) * C + c];
}

// ---------------------------------------------------------------------------
// Stage 2: per-head projection GEMM.  32(M) x 64(N=HD) tile per wave,
// register double-buffered over K.
//   X:  [B, S, E] f16 row-major
//   WT: [H, HD, E] f16   (BT layout)
//   Out: transposed==0 -> [B,H,S,HD], transposed==1 -> [B,H,HD,S]
// ---------------------------------------------------------------------------
__global__ void proj_kernel(const _Float16* __restrict__ X,
                            const _Float16* __restrict__ WT,
                            _Float16* __restrict__ Out,
                            int S, int transposed) {
  const int lane = threadIdx.x;
  const int m0 = blockIdx.x * 32;
  const int h  = blockIdx.y;
  const int b  = blockIdx.z;
  const _Float16* A0 = X + ((size_t)b * S + m0) * E_DIM;
  const _Float16* A1 = A0 + (size_t)16 * E_DIM;
  const _Float16* Wh = WT + (size_t)h * HD_DIM * E_DIM;

  v8f acc[2][4] = {};

  // prime the pipeline
  v16h ca0 = ld_afrag(A0, E_DIM, lane);
  v16h ca1 = ld_afrag(A1, E_DIM, lane);
  v16h cb[4];
#pragma unroll
  for (int nt = 0; nt < 4; ++nt)
    cb[nt] = ld_bfrag(Wh + (size_t)(nt * 16) * E_DIM, E_DIM, lane);

  for (int k0 = 0; k0 < E_DIM; k0 += 32) {
    const int kn = (k0 + 32 < E_DIM) ? k0 + 32 : k0;  // clamped prefetch
    v16h na0 = ld_afrag(A0 + kn, E_DIM, lane);
    v16h na1 = ld_afrag(A1 + kn, E_DIM, lane);
    v16h nb[4];
#pragma unroll
    for (int nt = 0; nt < 4; ++nt)
      nb[nt] = ld_bfrag(Wh + (size_t)(nt * 16) * E_DIM + kn, E_DIM, lane);

#pragma unroll
    for (int nt = 0; nt < 4; ++nt) {
      acc[0][nt] = wmma_f16(ca0, cb[nt], acc[0][nt]);
      acc[1][nt] = wmma_f16(ca1, cb[nt], acc[1][nt]);
    }
    ca0 = na0; ca1 = na1;
#pragma unroll
    for (int nt = 0; nt < 4; ++nt) cb[nt] = nb[nt];
  }

  // C layout: VGPR r -> row base + r + 8*(lane>>4), column = lane&15
  const int half = lane >> 4, nn = lane & 15;
#pragma unroll
  for (int mt = 0; mt < 2; ++mt)
#pragma unroll
    for (int nt = 0; nt < 4; ++nt)
#pragma unroll
      for (int r = 0; r < 8; ++r) {
        int row = m0 + mt * 16 + r + 8 * half;
        int col = nt * 16 + nn;
        _Float16 v = (_Float16)acc[mt][nt][r];
        if (!transposed)
          Out[(((size_t)b * H_NUM + h) * S + row) * HD_DIM + col] = v;
        else
          Out[(((size_t)b * H_NUM + h) * HD_DIM + col) * S + row] = v;
      }
}

// ---------------------------------------------------------------------------
// Stage 3: flash attention. One wave per 16-row q tile per (b,h).
// K fragments double-buffered across key blocks; V fragments issued early so
// they land during the softmax VALU work.
//   Q:  [B,H,SCHI,HD] f16   K: [B,H,SENG,HD] f16   VT: [B,H,HD,SENG] f16
//   O:  [B,SCHI,E] f16  (head-concatenated, feeds the out-projection GEMM)
// ---------------------------------------------------------------------------
__global__ void attn_kernel(const _Float16* __restrict__ Q,
                            const _Float16* __restrict__ K,
                            const _Float16* __restrict__ VT,
                            _Float16* __restrict__ O) {
  __shared__ _Float16 pLds[16 * 32];   // P tile staging: C-layout -> A-layout
  const int lane = threadIdx.x;
  const int q0 = blockIdx.x * 16;
  const int h  = blockIdx.y;
  const int b  = blockIdx.z;

  const _Float16* Qbh = Q + (((size_t)b * H_NUM + h) * SCHI + q0) * HD_DIM;
  const _Float16* Kbh = K + ((size_t)b * H_NUM + h) * SENG * HD_DIM;
  const _Float16* Vbh = VT + ((size_t)b * H_NUM + h) * HD_DIM * SENG;

  const v16h qa0 = ld_afrag(Qbh + 0,  HD_DIM, lane);   // d = 0..31
  const v16h qa1 = ld_afrag(Qbh + 32, HD_DIM, lane);   // d = 32..63

  v8f o[4] = {};
  float mrow[8], srow[8];
#pragma unroll
  for (int r = 0; r < 8; ++r) { mrow[r] = -3.0e38f; srow[r] = 0.f; }
  const float scale = 0.125f;  // 1/sqrt(64)
  const int half = lane >> 4, nn = lane & 15;

  // prime K fragments for block n0 = 0
  v16h ck[4];
  ck[0] = ld_bfrag(Kbh + 0,                        HD_DIM, lane); // keys 0..15,  d 0..31
  ck[1] = ld_bfrag(Kbh + 32,                       HD_DIM, lane); // keys 0..15,  d 32..63
  ck[2] = ld_bfrag(Kbh + (size_t)16 * HD_DIM + 0,  HD_DIM, lane); // keys 16..31, d 0..31
  ck[3] = ld_bfrag(Kbh + (size_t)16 * HD_DIM + 32, HD_DIM, lane); // keys 16..31, d 32..63

  for (int n0 = 0; n0 < SENG; n0 += 32) {
    // V fragments for this block: independent of scores, issue early
    v16h vb[4];
#pragma unroll
    for (int dt = 0; dt < 4; ++dt)
      vb[dt] = ld_bfrag(Vbh + (size_t)(dt * 16) * SENG + n0, SENG, lane);

    // S = Q K^T for 32 keys: two 16-wide score tiles
    v8f s0 = {}, s1 = {};
    s0 = wmma_f16(qa0, ck[0], s0);
    s0 = wmma_f16(qa1, ck[1], s0);
    s1 = wmma_f16(qa0, ck[2], s1);
    s1 = wmma_f16(qa1, ck[3], s1);

    // prefetch K fragments for the next block; fly during softmax
    const int n1 = (n0 + 32 < SENG) ? n0 + 32 : n0;
    v16h nk[4];
    nk[0] = ld_bfrag(Kbh + (size_t)n1 * HD_DIM + 0,         HD_DIM, lane);
    nk[1] = ld_bfrag(Kbh + (size_t)n1 * HD_DIM + 32,        HD_DIM, lane);
    nk[2] = ld_bfrag(Kbh + (size_t)(n1 + 16) * HD_DIM + 0,  HD_DIM, lane);
    nk[3] = ld_bfrag(Kbh + (size_t)(n1 + 16) * HD_DIM + 32, HD_DIM, lane);

    // online softmax; row M = r + 8*half lives in 16 lanes of this half-wave
#pragma unroll
    for (int r = 0; r < 8; ++r) {
      float a  = s0[r] * scale;
      float bv = s1[r] * scale;
      float mx = fmaxf(a, bv);
#pragma unroll
      for (int off = 1; off < 16; off <<= 1)
        mx = fmaxf(mx, __shfl_xor(mx, off, 32));
      float mnew  = fmaxf(mrow[r], mx);
      float alpha = __expf(mrow[r] - mnew);
      mrow[r] = mnew;
      float p0 = __expf(a - mnew);
      float p1 = __expf(bv - mnew);
      float ps = p0 + p1;
#pragma unroll
      for (int off = 1; off < 16; off <<= 1)
        ps += __shfl_xor(ps, off, 32);
      srow[r] = srow[r] * alpha + ps;
      o[0][r] *= alpha; o[1][r] *= alpha; o[2][r] *= alpha; o[3][r] *= alpha;
      int row = r + 8 * half;
      pLds[row * 32 + nn]      = (_Float16)p0;
      pLds[row * 32 + 16 + nn] = (_Float16)p1;
    }
    __syncthreads();  // order C-layout stores vs A-layout loads (1-wave WG)

    // reload P as an A-fragment (16x32 f16); same-wave LDS ops are in-order
    const int arow = lane & 15;
    const int akb  = (lane >> 4) * 8;
    v16h pa;
#pragma unroll
    for (int i = 0; i < 8; ++i) {
      pa[i]     = pLds[arow * 32 + akb + i];
      pa[i + 8] = pLds[arow * 32 + akb + 16 + i];
    }

    // O += P @ V   (VT rows are d, contiguous over keys)
#pragma unroll
    for (int dt = 0; dt < 4; ++dt)
      o[dt] = wmma_f16(pa, vb[dt], o[dt]);

#pragma unroll
    for (int kk = 0; kk < 4; ++kk) ck[kk] = nk[kk];
  }

  // normalize + store head-concatenated f16
#pragma unroll
  for (int dt = 0; dt < 4; ++dt)
#pragma unroll
    for (int r = 0; r < 8; ++r) {
      int row = q0 + r + 8 * half;
      int col = h * HD_DIM + dt * 16 + nn;
      O[((size_t)b * SCHI + row) * E_DIM + col] = (_Float16)(o[dt][r] / srow[r]);
    }
}

// ---------------------------------------------------------------------------
// Stage 4: output projection  Out[BS,E] = Att[BS,E] @ Wo[E,E]  (f32 result)
// 32x64 tile per wave, double-buffered over K.
// ---------------------------------------------------------------------------
__global__ void outproj_kernel(const _Float16* __restrict__ A,
                               const _Float16* __restrict__ WoT, // [N=E][K=E]
                               float* __restrict__ Out) {
  const int lane = threadIdx.x;
  const int m0 = blockIdx.x * 32;
  const int n0 = blockIdx.y * 64;
  const _Float16* A0 = A + (size_t)m0 * E_DIM;
  const _Float16* A1 = A0 + (size_t)16 * E_DIM;
  const _Float16* W0 = WoT + (size_t)n0 * E_DIM;

  v8f acc[2][4] = {};
  v16h ca0 = ld_afrag(A0, E_DIM, lane);
  v16h ca1 = ld_afrag(A1, E_DIM, lane);
  v16h cb[4];
#pragma unroll
  for (int nt = 0; nt < 4; ++nt)
    cb[nt] = ld_bfrag(W0 + (size_t)(nt * 16) * E_DIM, E_DIM, lane);

  for (int k0 = 0; k0 < E_DIM; k0 += 32) {
    const int kn = (k0 + 32 < E_DIM) ? k0 + 32 : k0;
    v16h na0 = ld_afrag(A0 + kn, E_DIM, lane);
    v16h na1 = ld_afrag(A1 + kn, E_DIM, lane);
    v16h nb[4];
#pragma unroll
    for (int nt = 0; nt < 4; ++nt)
      nb[nt] = ld_bfrag(W0 + (size_t)(nt * 16) * E_DIM + kn, E_DIM, lane);

#pragma unroll
    for (int nt = 0; nt < 4; ++nt) {
      acc[0][nt] = wmma_f16(ca0, cb[nt], acc[0][nt]);
      acc[1][nt] = wmma_f16(ca1, cb[nt], acc[1][nt]);
    }
    ca0 = na0; ca1 = na1;
#pragma unroll
    for (int nt = 0; nt < 4; ++nt) cb[nt] = nb[nt];
  }

  const int half = lane >> 4, nn = lane & 15;
#pragma unroll
  for (int mt = 0; mt < 2; ++mt)
#pragma unroll
    for (int nt = 0; nt < 4; ++nt)
#pragma unroll
      for (int r = 0; r < 8; ++r)
        Out[(size_t)(m0 + mt * 16 + r + 8 * half) * E_DIM + (n0 + nt * 16 + nn)] =
            acc[mt][nt][r];
}

// ---------------------------------------------------------------------------
extern "C" void kernel_launch(void* const* d_in, const int* in_sizes, int n_in,
                              void* d_out, int out_size, void* d_ws, size_t ws_size,
                              hipStream_t stream) {
  const float* eng = (const float*)d_in[0];  // [B,SENG,E]
  const float* chi = (const float*)d_in[1];  // [B,SCHI,E]
  const float* Wq  = (const float*)d_in[2];  // [H,E,HD]
  const float* Wk  = (const float*)d_in[3];
  const float* Wv  = (const float*)d_in[4];
  const float* Wo  = (const float*)d_in[5];  // [E,E]
  float* out = (float*)d_out;                // [B,SCHI,E]

  char* ws = (char*)d_ws;
  size_t off = 0;
  auto alloc = [&](size_t bytes) { char* p = ws + off; off += bytes; return p; };

  const size_t n_eng = (size_t)B_NUM * SENG * E_DIM;
  const size_t n_chi = (size_t)B_NUM * SCHI * E_DIM;
  const size_t n_w   = (size_t)H_NUM * E_DIM * HD_DIM;

  _Float16* eng_h = (_Float16*)alloc(n_eng * 2);
  _Float16* chi_h = (_Float16*)alloc(n_chi * 2);
  _Float16* WqT   = (_Float16*)alloc(n_w * 2);
  _Float16* WkT   = (_Float16*)alloc(n_w * 2);
  _Float16* WvT   = (_Float16*)alloc(n_w * 2);
  _Float16* WoT   = (_Float16*)alloc((size_t)E_DIM * E_DIM * 2);
  _Float16* qb    = (_Float16*)alloc((size_t)B_NUM * H_NUM * SCHI * HD_DIM * 2);
  _Float16* kb    = (_Float16*)alloc((size_t)B_NUM * H_NUM * SENG * HD_DIM * 2);
  _Float16* vTb   = (_Float16*)alloc((size_t)B_NUM * H_NUM * HD_DIM * SENG * 2);
  _Float16* attb  = (_Float16*)alloc((size_t)B_NUM * SCHI * E_DIM * 2);
  (void)ws_size;

  // Stage 1: convert activations, transpose+convert weights
  {
    size_t n4 = n_eng / 4;
    cvt_f32_to_f16<<<dim3((unsigned)((n4 + 255) / 256)), 256, 0, stream>>>(eng, eng_h, n4);
    n4 = n_chi / 4;
    cvt_f32_to_f16<<<dim3((unsigned)((n4 + 255) / 256)), 256, 0, stream>>>(chi, chi_h, n4);

    dim3 tb(16, 16);
    dim3 tgW(HD_DIM / 16, E_DIM / 16, H_NUM);
    transpose_cvt<<<tgW, tb, 0, stream>>>(Wq, WqT, E_DIM, HD_DIM);
    transpose_cvt<<<tgW, tb, 0, stream>>>(Wk, WkT, E_DIM, HD_DIM);
    transpose_cvt<<<tgW, tb, 0, stream>>>(Wv, WvT, E_DIM, HD_DIM);
    dim3 tgO(E_DIM / 16, E_DIM / 16, 1);
    transpose_cvt<<<tgO, tb, 0, stream>>>(Wo, WoT, E_DIM, E_DIM);
  }

  // Stage 2: projections (q row-major, k row-major, v transposed)
  proj_kernel<<<dim3(SCHI / 32, H_NUM, B_NUM), 32, 0, stream>>>(chi_h, WqT, qb, SCHI, 0);
  proj_kernel<<<dim3(SENG / 32, H_NUM, B_NUM), 32, 0, stream>>>(eng_h, WkT, kb, SENG, 0);
  proj_kernel<<<dim3(SENG / 32, H_NUM, B_NUM), 32, 0, stream>>>(eng_h, WvT, vTb, SENG, 1);

  // Stage 3: flash attention
  attn_kernel<<<dim3(SCHI / 16, H_NUM, B_NUM), 32, 0, stream>>>(qb, kb, vTb, attb);

  // Stage 4: output projection -> f32 d_out
  outproj_kernel<<<dim3((B_NUM * SCHI) / 32, E_DIM / 64), 32, 0, stream>>>(attb, WoT, out);
  (void)in_sizes; (void)n_in; (void)out_size;
}
